// Agent_lstm_56410100466270
// MI455X (gfx1250) — matrix operand order repaired and checked by reading, not compiled, not run on hardware
//
#include <hip/hip_runtime.h>
#include <hip/hip_bf16.h>

#define T_STEPS 256
#define BATCH   4096
#define HID     64
#define NW      4            // waves per block
#define LANES   32

typedef __attribute__((ext_vector_type(16))) __bf16 v16bf;
typedef __attribute__((ext_vector_type(8)))  float  v8f;

union Frag {
    v16bf          bf;
    unsigned short us[16];
    unsigned int   u32[8];
};

__device__ __forceinline__ unsigned short f2bf(float f) {
    union { float f; unsigned int u; } v; v.f = f;
    unsigned int u = v.u;
    return (unsigned short)((u + 0x7FFFu + ((u >> 16) & 1u)) >> 16);
}

__device__ __forceinline__ float sigm(float x) { return 1.0f / (1.0f + __expf(-x)); }
__device__ __forceinline__ float tanh_(float x) { return 1.0f - 2.0f / (__expf(2.0f * x) + 1.0f); }

__device__ __forceinline__ v8f wmma_bf16(v16bf a, v16bf b, v8f c) {
    // emits v_wmma_f32_16x16x32_bf16
    return __builtin_amdgcn_wmma_f32_16x16x32_bf16(
        /*neg_a=*/false, a, /*neg_b=*/false, b,
        /*c_mod=*/(short)0, c, /*reuse_a=*/false, /*reuse_b=*/false);
}

// B fragment: W stored in LDS as [N][Ktot] row-major bf16 (i.e. transposed weight).
// VGPR j holds K pair at 2*(j&3) + 8*(lane>=16) + 16*(j>=4).
__device__ __forceinline__ v16bf load_bfrag(const unsigned short* W, int Ktot,
                                            int N0, int K0, int lane) {
    Frag a;
    const unsigned int* p = (const unsigned int*)
        (W + (N0 + (lane & 15)) * Ktot + K0 + ((lane >> 4) << 3));
#pragma unroll
    for (int j = 0; j < 8; ++j) a.u32[j] = p[(j & 3) + ((j >> 2) << 3)];
    return a.bf;
}

// A fragment: M-tile stored in LDS as [16][Ktot] row-major bf16.
// Lane row = lane&15; VGPR j holds K pair at 2*(j&3) + 8*(lane>=16) + 16*(j>=4).
__device__ __forceinline__ v16bf load_afrag(const unsigned short* Mt, int Ktot,
                                            int K0, int lane, bool zero) {
    Frag a;
    int row = lane & 15;
    int kb  = K0 + ((lane >> 4) << 3);
#pragma unroll
    for (int j = 0; j < 8; ++j) {
        int k = kb + ((j & 3) << 1) + ((j >> 2) << 4);
        a.u32[j] = zero ? 0u : *(const unsigned int*)(Mt + row * Ktot + k);
    }
    return a.bf;
}

// A fragment for x rows (K=9, zero-padded to 32), loaded straight from global.
__device__ __forceinline__ v16bf load_ax(const float* __restrict__ x, int rowbase, int lane) {
    Frag a;
    int kb = (lane >> 4) << 3;
    const float* xr = x + (rowbase + (lane & 15)) * 9;
#pragma unroll
    for (int j = 0; j < 8; ++j) {
        int k = kb + ((j & 3) << 1) + ((j >> 2) << 4);
        float f0 = (k     < 9) ? xr[k]     : 0.0f;
        float f1 = (k + 1 < 9) ? xr[k + 1] : 0.0f;
        a.us[2 * j]     = f2bf(f0);
        a.us[2 * j + 1] = f2bf(f1);
    }
    return a.bf;
}

// Store a C/D-layout f32 accumulator (one 16x16 N-tile) into a bf16 LDS tile [16][ld].
__device__ __forceinline__ void store_cd(unsigned short* tile, int ld, int nt,
                                         int lane, v8f acc) {
    int col = (lane & 15) + (nt << 4);
    int rb  = (lane >> 4) << 3;
#pragma unroll
    for (int i = 0; i < 8; ++i) tile[(rb + i) * ld + col] = f2bf(acc[i]);
}

__global__ __launch_bounds__(NW * LANES)
void agent_lstm_kernel(const float* __restrict__ x,    const int* __restrict__ done,
                       const float* __restrict__ h0,   const float* __restrict__ c0,
                       const float* __restrict__ Wr1,  const float* __restrict__ br1,
                       const float* __restrict__ Wr2,  const float* __restrict__ br2,
                       const float* __restrict__ Wih,  const float* __restrict__ Whh,
                       const float* __restrict__ b_lstm,
                       const float* __restrict__ Wc1,  const float* __restrict__ bc1,
                       const float* __restrict__ Wc2,  const float* __restrict__ bc2,
                       const float* __restrict__ Wc3,  const float* __restrict__ bc3,
                       float* __restrict__ out) {
    // ---- shared weight staging (bf16, transposed [N][K] for B-fragment reads) ----
    __shared__ unsigned short s_wr1[32 * 32];     // [N=32][K=32] (K padded from 9)
    __shared__ unsigned short s_wr2[64 * 32];     // [N=64][K=32]
    __shared__ unsigned short s_wih[256 * 64];    // [N=256][K=64]
    __shared__ unsigned short s_whh[256 * 64];    // [N=256][K=64]
    __shared__ unsigned short s_wc1[16 * 64];     // [N=16][K=64]
    __shared__ float s_br1[32], s_br2[64], s_bl[256], s_bc1[16], s_bc2[8];
    __shared__ float s_wc2[16 * 8], s_wc3[8], s_bc3[1];
    // per-wave scratch tiles
    __shared__ unsigned short s_t1[NW][16 * 32];  // relu(x@Wr1) bf16
    __shared__ unsigned short s_hid[NW][16 * 64]; // root MLP out bf16
    __shared__ unsigned short s_h[NW][16 * 64];   // lstm hidden bf16
    __shared__ float          s_c1[NW][16 * 16];  // critic layer1 f32

    const int tid = threadIdx.x, nthr = blockDim.x;
    for (int i = tid; i < 256 * 64; i += nthr) {
        int n = i >> 6, k = i & 63;
        s_wih[i] = f2bf(Wih[k * 256 + n]);
        s_whh[i] = f2bf(Whh[k * 256 + n]);
    }
    for (int i = tid; i < 32 * 32; i += nthr) {
        int n = i >> 5, k = i & 31;
        s_wr1[i] = f2bf(k < 9 ? Wr1[k * 32 + n] : 0.0f);
    }
    for (int i = tid; i < 64 * 32; i += nthr) {
        int n = i >> 5, k = i & 31;
        s_wr2[i] = f2bf(Wr2[k * 64 + n]);
    }
    for (int i = tid; i < 16 * 64; i += nthr) {
        int n = i >> 6, k = i & 63;
        s_wc1[i] = f2bf(Wc1[k * 16 + n]);
    }
    for (int i = tid; i < 256; i += nthr) {
        s_bl[i] = b_lstm[i];
        if (i < 32)  s_br1[i] = br1[i];
        if (i < 64)  s_br2[i] = br2[i];
        if (i < 16)  s_bc1[i] = bc1[i];
        if (i < 128) s_wc2[i] = Wc2[i];
        if (i < 8)  { s_bc2[i] = bc2[i]; s_wc3[i] = Wc3[i]; }
        if (i == 0)  s_bc3[0] = bc3[0];
    }
    __syncthreads();

    const int wave = tid >> 5, lane = tid & 31;
    const int tile = blockIdx.x * NW + wave;
    const int b0   = tile << 4;              // 16 batch rows per wave
    const int rb   = (lane >> 4) << 3;       // C/D row base for this lane
    const int colb = lane & 15;              // C/D col within N-tile

    unsigned short* t1t  = s_t1[wave];
    unsigned short* hidt = s_hid[wave];
    unsigned short* ht   = s_h[wave];
    float*          c1t  = s_c1[wave];

    // ---- init state: c in C/D accumulator layout, h as bf16 LDS tile ----
    v8f c[4];
#pragma unroll
    for (int jb = 0; jb < 4; ++jb)
#pragma unroll
        for (int i = 0; i < 8; ++i)
            c[jb][i] = c0[(b0 + rb + i) * HID + colb + (jb << 4)];
    {
        int hrow = lane & 15, cstart = (lane >> 4) * 32;
        for (int kk = 0; kk < 32; ++kk)
            ht[hrow * HID + cstart + kk] = f2bf(h0[(b0 + hrow) * HID + cstart + kk]);
    }

    for (int t = 0; t < T_STEPS; ++t) {
        const int base = t * BATCH + b0;
        // per-row episode reset: h masked at A-fragment read, c masked elementwise
        const int myd = done[base + (lane & 15)];
        v16bf ah0 = load_afrag(ht, 64, 0,  lane, myd != 0);
        v16bf ah1 = load_afrag(ht, 64, 32, lane, myd != 0);
        float dn = (float)myd;
        float m[8];
#pragma unroll
        for (int i = 0; i < 8; ++i) m[i] = 1.0f - __shfl(dn, rb + i);
#pragma unroll
        for (int jb = 0; jb < 4; ++jb)
#pragma unroll
            for (int i = 0; i < 8; ++i) c[jb][i] *= m[i];

        // ---- root MLP layer 1: relu(x @ Wr1 + br1), K padded 9->32 ----
        v16bf ax = load_ax(x, base, lane);
#pragma unroll
        for (int nt = 0; nt < 2; ++nt) {
            v8f a1;
            float bb = s_br1[colb + (nt << 4)];
#pragma unroll
            for (int i = 0; i < 8; ++i) a1[i] = bb;
            a1 = wmma_bf16(ax, load_bfrag(s_wr1, 32, nt << 4, 0, lane), a1);
#pragma unroll
            for (int i = 0; i < 8; ++i) a1[i] = fmaxf(a1[i], 0.0f);
            store_cd(t1t, 32, nt, lane, a1);
        }
        v16bf at1 = load_afrag(t1t, 32, 0, lane, false);

        // ---- root MLP layer 2: relu(t1 @ Wr2 + br2) -> hidden [16,64] ----
#pragma unroll
        for (int nt = 0; nt < 4; ++nt) {
            v8f a2;
            float bb = s_br2[colb + (nt << 4)];
#pragma unroll
            for (int i = 0; i < 8; ++i) a2[i] = bb;
            a2 = wmma_bf16(at1, load_bfrag(s_wr2, 32, nt << 4, 0, lane), a2);
#pragma unroll
            for (int i = 0; i < 8; ++i) a2[i] = fmaxf(a2[i], 0.0f);
            store_cd(hidt, 64, nt, lane, a2);
        }
        v16bf ahid0 = load_afrag(hidt, 64, 0,  lane, false);
        v16bf ahid1 = load_afrag(hidt, 64, 32, lane, false);

        // ---- gates: g = b_lstm + hidden@Wih + h@Whh ; order i,f,g,o ----
#pragma unroll
        for (int jb = 0; jb < 4; ++jb) {            // hidden-column block (16 cols)
            v8f g[4];
#pragma unroll
            for (int gg = 0; gg < 4; ++gg) {        // gate
                int nt = (gg << 2) + jb;            // N-tile inside 4H=256
                float bb = s_bl[colb + (nt << 4)];
#pragma unroll
                for (int i = 0; i < 8; ++i) g[gg][i] = bb;
                g[gg] = wmma_bf16(ahid0, load_bfrag(s_wih, 64, nt << 4, 0,  lane), g[gg]);
                g[gg] = wmma_bf16(ahid1, load_bfrag(s_wih, 64, nt << 4, 32, lane), g[gg]);
                g[gg] = wmma_bf16(ah0,   load_bfrag(s_whh, 64, nt << 4, 0,  lane), g[gg]);
                g[gg] = wmma_bf16(ah1,   load_bfrag(s_whh, 64, nt << 4, 32, lane), g[gg]);
            }
            v8f hnew;
#pragma unroll
            for (int i = 0; i < 8; ++i) {
                float ci = sigm(g[1][i]) * c[jb][i] + sigm(g[0][i]) * tanh_(g[2][i]);
                c[jb][i] = ci;
                hnew[i]  = sigm(g[3][i]) * tanh_(ci);
            }
            store_cd(ht, 64, jb, lane, hnew);       // persist hidden for critic + next step
        }

        // ---- critic head: tanh(h@Wc1+bc1) via WMMA, then VALU tail ----
        v16bf an0 = load_afrag(ht, 64, 0,  lane, false);
        v16bf an1 = load_afrag(ht, 64, 32, lane, false);
        v8f v1;
        {
            float bb = s_bc1[colb];
#pragma unroll
            for (int i = 0; i < 8; ++i) v1[i] = bb;
        }
        v1 = wmma_bf16(an0, load_bfrag(s_wc1, 64, 0, 0,  lane), v1);
        v1 = wmma_bf16(an1, load_bfrag(s_wc1, 64, 0, 32, lane), v1);
#pragma unroll
        for (int i = 0; i < 8; ++i) c1t[(rb + i) * 16 + colb] = tanh_(v1[i]);

        if (lane < 16) {                            // one row per lane: 16x8, 8x1 tail
            float vv = s_bc3[0];
#pragma unroll
            for (int o = 0; o < 8; ++o) {
                float acc = s_bc2[o];
#pragma unroll
                for (int k = 0; k < 16; ++k) acc += c1t[lane * 16 + k] * s_wc2[k * 8 + o];
                vv += tanh_(acc) * s_wc3[o];
            }
            out[base + lane] = vv;
        }

        // prefetch next timestep's x rows into cache (global_prefetch_b8)
        if (t + 1 < T_STEPS)
            __builtin_prefetch(x + (base + BATCH + (lane & 15)) * 9, 0, 1);
    }
}

extern "C" void kernel_launch(void* const* d_in, const int* in_sizes, int n_in,
                              void* d_out, int out_size, void* d_ws, size_t ws_size,
                              hipStream_t stream) {
    (void)in_sizes; (void)n_in; (void)d_ws; (void)ws_size; (void)out_size;
    const float* x      = (const float*)d_in[0];
    const int*   done   = (const int*)  d_in[1];
    const float* h0     = (const float*)d_in[2];
    const float* c0     = (const float*)d_in[3];
    const float* Wr1    = (const float*)d_in[4];
    const float* br1    = (const float*)d_in[5];
    const float* Wr2    = (const float*)d_in[6];
    const float* br2    = (const float*)d_in[7];
    const float* Wih    = (const float*)d_in[8];
    const float* Whh    = (const float*)d_in[9];
    const float* b_lstm = (const float*)d_in[10];
    const float* Wc1    = (const float*)d_in[11];
    const float* bc1    = (const float*)d_in[12];
    const float* Wc2    = (const float*)d_in[13];
    const float* bc2    = (const float*)d_in[14];
    const float* Wc3    = (const float*)d_in[15];
    const float* bc3    = (const float*)d_in[16];
    float* out = (float*)d_out;

    dim3 grid(BATCH / (16 * NW));   // 64 blocks
    dim3 block(NW * LANES);         // 128 threads = 4 waves
    agent_lstm_kernel<<<grid, block, 0, stream>>>(x, done, h0, c0, Wr1, br1, Wr2, br2,
                                                  Wih, Whh, b_lstm, Wc1, bc1, Wc2, bc2,
                                                  Wc3, bc3, out);
}
